// Block_46110768890097
// MI455X (gfx1250) — compile-verified, matrix operand order checked by reading, not run on hardware
//
#include <hip/hip_runtime.h>
#include <math.h>

typedef unsigned int u32;
typedef __bf16 bf16;
typedef u32   v4u   __attribute__((ext_vector_type(4)));
typedef float v8f   __attribute__((ext_vector_type(8)));
typedef bf16  v16bf __attribute__((ext_vector_type(16)));

union Frag { v16bf v; v4u q[2]; };

__device__ __forceinline__ v8f zero8() {
    v8f z;
#pragma unroll
    for (int i = 0; i < 8; ++i) z[i] = 0.f;
    return z;
}

__device__ __forceinline__ v8f wmma_bf16(v16bf a, v16bf b, v8f c) {
    return __builtin_amdgcn_wmma_f32_16x16x32_bf16(false, a, false, b, (short)0, c, false, false);
}

// ---------------------------------------------------------------------------
// float -> bf16 conversion (weights)
// ---------------------------------------------------------------------------
__global__ void cvt_f32_to_bf16(const float* __restrict__ s, bf16* __restrict__ d, int n) {
    int i = blockIdx.x * blockDim.x + threadIdx.x;
    int stride = gridDim.x * blockDim.x;
    for (; i < n; i += stride) d[i] = (bf16)s[i];
}

// ---------------------------------------------------------------------------
// LayerNorm over C=768, one block per row, bf16 output
// ---------------------------------------------------------------------------
__global__ __launch_bounds__(256) void ln_768(const float* __restrict__ x,
                                              const float* __restrict__ g,
                                              const float* __restrict__ bta,
                                              bf16* __restrict__ out) {
    __shared__ float red[256];
    const int row = blockIdx.x, t = threadIdx.x;
    const float* xr = x + (size_t)row * 768;
    float v0 = xr[t], v1 = xr[t + 256], v2 = xr[t + 512];
    red[t] = v0 + v1 + v2;
    __syncthreads();
    for (int o = 128; o > 0; o >>= 1) { if (t < o) red[t] += red[t + o]; __syncthreads(); }
    float mu = red[0] * (1.f / 768.f);
    __syncthreads();
    float d0 = v0 - mu, d1 = v1 - mu, d2 = v2 - mu;
    red[t] = d0 * d0 + d1 * d1 + d2 * d2;
    __syncthreads();
    for (int o = 128; o > 0; o >>= 1) { if (t < o) red[t] += red[t + o]; __syncthreads(); }
    float rstd = rsqrtf(red[0] * (1.f / 768.f) + 1e-5f);
    bf16* orow = out + (size_t)row * 768;
    orow[t]       = (bf16)(d0 * rstd * g[t]       + bta[t]);
    orow[t + 256] = (bf16)(d1 * rstd * g[t + 256] + bta[t + 256]);
    orow[t + 512] = (bf16)(d2 * rstd * g[t + 512] + bta[t + 512]);
}

// ---------------------------------------------------------------------------
// Generic WMMA GEMM: out[M,Ncol] = A[M,K](bf16) @ W[Ncol,K](bf16)^T  (+epilogue)
// EPI 0: plain -> bf16 out
// EPI 1: +bias +residual(f32) -> f32 out
// EPI 2: +bias, exact GELU -> bf16 out
// Block: 256 threads (8 waves). Tile 128x128, K-step 32. Dims assumed divisible.
// ---------------------------------------------------------------------------
template <int EPI>
__global__ __launch_bounds__(256) void gemm_bf16_wmma(
    const bf16* __restrict__ A, const bf16* __restrict__ W,
    const float* __restrict__ bias, const float* __restrict__ res,
    void* __restrict__ outv, int Ncol, int K) {
    __shared__ bf16 As[128][32];
    __shared__ bf16 Ws[128][32];
    const int t = threadIdx.x, lane = t & 31, wv = t >> 5;
    const int wr = wv >> 1, wc = wv & 1;
    const int l16 = lane & 15, hsel = lane >> 4;
    const int rowBase = blockIdx.y * 128, colBase = blockIdx.x * 128;

    v8f acc[2][4];
#pragma unroll
    for (int mt = 0; mt < 2; ++mt)
#pragma unroll
        for (int nt = 0; nt < 4; ++nt) acc[mt][nt] = zero8();

    for (int k0 = 0; k0 < K; k0 += 32) {
        __syncthreads();
#pragma unroll
        for (int i = 0; i < 2; ++i) {
            int c = t * 2 + i;
            int r = c >> 2, o8 = (c & 3) * 8;
            *(v4u*)&As[r][o8] = *(const v4u*)(A + (size_t)(rowBase + r) * K + k0 + o8);
            *(v4u*)&Ws[r][o8] = *(const v4u*)(W + (size_t)(colBase + r) * K + k0 + o8);
        }
        if (k0 + 32 < K) {
            __builtin_prefetch(A + (size_t)(rowBase + (t >> 1)) * K + k0 + 32, 0, 0);
            __builtin_prefetch(W + (size_t)(colBase + (t >> 1)) * K + k0 + 32, 0, 0);
        }
        __syncthreads();
        Frag af[2];
#pragma unroll
        for (int mt = 0; mt < 2; ++mt) {
            int rrow = wr * 32 + mt * 16 + l16;
            af[mt].q[0] = *(const v4u*)&As[rrow][hsel * 8];
            af[mt].q[1] = *(const v4u*)&As[rrow][hsel * 8 + 16];
        }
#pragma unroll
        for (int nt = 0; nt < 4; ++nt) {
            int rcol = wc * 64 + nt * 16 + l16;
            Frag bfr;
            bfr.q[0] = *(const v4u*)&Ws[rcol][hsel * 8];
            bfr.q[1] = *(const v4u*)&Ws[rcol][hsel * 8 + 16];
#pragma unroll
            for (int mt = 0; mt < 2; ++mt)
                acc[mt][nt] = wmma_bf16(af[mt].v, bfr.v, acc[mt][nt]);
        }
    }

#pragma unroll
    for (int mt = 0; mt < 2; ++mt) {
#pragma unroll
        for (int nt = 0; nt < 4; ++nt) {
            int gcol = colBase + wc * 64 + nt * 16 + l16;
            float bi = (EPI != 0) ? bias[gcol] : 0.f;
#pragma unroll
            for (int r = 0; r < 8; ++r) {
                int grow = rowBase + wr * 32 + mt * 16 + hsel * 8 + r;
                float val = acc[mt][nt][r] + bi;
                size_t oidx = (size_t)grow * Ncol + gcol;
                if (EPI == 1) {
                    ((float*)outv)[oidx] = val + res[oidx];
                } else if (EPI == 2) {
                    val = 0.5f * val * (1.f + erff(val * 0.70710678f));
                    ((bf16*)outv)[oidx] = (bf16)val;
                } else {
                    ((bf16*)outv)[oidx] = (bf16)val;
                }
            }
        }
    }
}

// ---------------------------------------------------------------------------
// Fused flash attention with relative-position bias.
// qkv: bf16 [B, N, 3C] (q at col h*64+d, k at C+..., v at 2C+...)
// Grid: (N/64, B*H), 128 threads (4 waves); each wave owns 16 query rows.
// ---------------------------------------------------------------------------
__global__ __launch_bounds__(128) void attn_fused(const bf16* __restrict__ qkv,
                                                  const float* __restrict__ rel_table,
                                                  bf16* __restrict__ aout) {
    constexpr int Nn = 1024, Cc = 768, Hh = 12, C3 = 3 * Cc;
    const int bh = blockIdx.y;
    const int b = bh / Hh, h = bh % Hh;
    const int t = threadIdx.x, lane = t & 31, wv = t >> 5;
    const int l16 = lane & 15, hsel = lane >> 4;
    const int rowbase = blockIdx.x * 64 + wv * 16;

    __shared__ bf16 VT[64][32];       // V^T tile: [d][n'] for current 32-col chunk
    __shared__ bf16 Pp[4][16][32];    // per-wave softmax tile in row-major

    const bf16* qp = qkv + (size_t)b * Nn * C3 + (size_t)h * 64;
    const bf16* kp = qp + Cc;
    const bf16* vp = qp + 2 * Cc;

    Frag qf[2];
#pragma unroll
    for (int kd = 0; kd < 2; ++kd) {
        const bf16* src = qp + (size_t)(rowbase + l16) * C3 + kd * 32 + hsel * 8;
        qf[kd].q[0] = *(const v4u*)src;
        qf[kd].q[1] = *(const v4u*)(src + 16);
    }

    v8f acc[4];
#pragma unroll
    for (int dt = 0; dt < 4; ++dt) acc[dt] = zero8();
    float mrow[8], lrow[8];
#pragma unroll
    for (int r = 0; r < 8; ++r) { mrow[r] = -1e30f; lrow[r] = 0.f; }

    for (int nc = 0; nc < Nn; nc += 32) {
        __syncthreads();  // all waves done reading previous VT
        {   // cooperative V^T staging: 128 threads x 16 elems
            int nn = t >> 2, d0 = (t & 3) * 16;
            const bf16* src = vp + (size_t)(nc + nn) * C3 + d0;
            union { v4u q[2]; bf16 e[16]; } vvld;
            vvld.q[0] = *(const v4u*)src;
            vvld.q[1] = *(const v4u*)(src + 8);
#pragma unroll
            for (int i = 0; i < 16; ++i) VT[d0 + i][nn] = vvld.e[i];
        }
        __syncthreads();

        // S = Q K^T for two 16-col tiles
        v8f s0t = zero8(), s1t = zero8();
#pragma unroll
        for (int kd = 0; kd < 2; ++kd) {
            Frag kf;
            const bf16* src0 = kp + (size_t)(nc + l16) * C3 + kd * 32 + hsel * 8;
            kf.q[0] = *(const v4u*)src0;
            kf.q[1] = *(const v4u*)(src0 + 16);
            s0t = wmma_bf16(qf[kd].v, kf.v, s0t);
            const bf16* src1 = kp + (size_t)(nc + 16 + l16) * C3 + kd * 32 + hsel * 8;
            kf.q[0] = *(const v4u*)src1;
            kf.q[1] = *(const v4u*)(src1 + 16);
            s1t = wmma_bf16(qf[kd].v, kf.v, s1t);
        }

        // bias + online softmax (row = hsel*8 + r, col = l16 within half-wave)
        float p0a[8], p1a[8], alpha[8];
#pragma unroll
        for (int r = 0; r < 8; ++r) {
            int mab = rowbase + hsel * 8 + r;
            int rel0 = min(max(nc + l16      - mab + 64, 0), 128);
            int rel1 = min(max(nc + 16 + l16 - mab + 64, 0), 128);
            float s0 = s0t[r] * 0.125f + rel_table[rel0 * Hh + h];
            float s1 = s1t[r] * 0.125f + rel_table[rel1 * Hh + h];
            float tm = fmaxf(s0, s1);
#pragma unroll
            for (int o = 1; o < 16; o <<= 1) tm = fmaxf(tm, __shfl_xor(tm, o, 32));
            float nm = fmaxf(mrow[r], tm);
            float p0 = __expf(s0 - nm), p1 = __expf(s1 - nm);
            float rs = p0 + p1;
#pragma unroll
            for (int o = 1; o < 16; o <<= 1) rs += __shfl_xor(rs, o, 32);
            float al = __expf(mrow[r] - nm);
            lrow[r] = lrow[r] * al + rs;
            mrow[r] = nm;
            alpha[r] = al;
            p0a[r] = p0;
            p1a[r] = p1;
        }
#pragma unroll
        for (int dt = 0; dt < 4; ++dt)
#pragma unroll
            for (int r = 0; r < 8; ++r) acc[dt][r] *= alpha[r];

        // P (C-layout) -> wave-private LDS row-major, re-read as A-fragment
#pragma unroll
        for (int r = 0; r < 8; ++r) {
            Pp[wv][hsel * 8 + r][l16]      = (bf16)p0a[r];
            Pp[wv][hsel * 8 + r][16 + l16] = (bf16)p1a[r];
        }
        Frag pf;
        pf.q[0] = *(const v4u*)&Pp[wv][l16][hsel * 8];
        pf.q[1] = *(const v4u*)&Pp[wv][l16][hsel * 8 + 16];
#pragma unroll
        for (int dt = 0; dt < 4; ++dt) {
            Frag vf;
            vf.q[0] = *(const v4u*)&VT[dt * 16 + l16][hsel * 8];
            vf.q[1] = *(const v4u*)&VT[dt * 16 + l16][hsel * 8 + 16];
            acc[dt] = wmma_bf16(pf.v, vf.v, acc[dt]);
        }
    }

#pragma unroll
    for (int dt = 0; dt < 4; ++dt) {
#pragma unroll
        for (int r = 0; r < 8; ++r) {
            int mab = rowbase + hsel * 8 + r;
            float val = acc[dt][r] / lrow[r];
            aout[((size_t)b * Nn + mab) * Cc + h * 64 + dt * 16 + l16] = (bf16)val;
        }
    }
}

// ---------------------------------------------------------------------------
extern "C" void kernel_launch(void* const* d_in, const int* in_sizes, int n_in,
                              void* d_out, int out_size, void* d_ws, size_t ws_size,
                              hipStream_t stream) {
    const float* x      = (const float*)d_in[0];
    const float* ln1_g  = (const float*)d_in[1];
    const float* ln1_b  = (const float*)d_in[2];
    const float* qkv_w  = (const float*)d_in[3];
    const float* rel_t  = (const float*)d_in[4];
    const float* proj_w = (const float*)d_in[5];
    const float* proj_b = (const float*)d_in[6];
    const float* ln2_g  = (const float*)d_in[7];
    const float* ln2_b  = (const float*)d_in[8];
    const float* fc1_w  = (const float*)d_in[9];
    const float* fc1_b  = (const float*)d_in[10];
    const float* fc2_w  = (const float*)d_in[11];
    const float* fc2_b  = (const float*)d_in[12];
    float* out = (float*)d_out;

    const size_t BN = (size_t)8 * 1024;
    char* ws = (char*)d_ws;
    size_t off = 0;
    auto alloc = [&](size_t bytes) -> void* {
        void* p = ws + off;
        off += (bytes + 255) & ~(size_t)255;
        return p;
    };
    bf16*  hbf   = (bf16*)alloc(BN * 768 * 2);
    bf16*  wqkv  = (bf16*)alloc((size_t)2304 * 768 * 2);
    bf16*  wproj = (bf16*)alloc((size_t)768 * 768 * 2);
    bf16*  wfc1  = (bf16*)alloc((size_t)3072 * 768 * 2);
    bf16*  wfc2  = (bf16*)alloc((size_t)768 * 3072 * 2);
    bf16*  qkvb  = (bf16*)alloc(BN * 2304 * 2);
    bf16*  aoutb = (bf16*)alloc(BN * 768 * 2);
    float* y1    = (float*)alloc(BN * 768 * 4);
    bf16*  h2    = (bf16*)alloc(BN * 768 * 2);
    bf16*  hbig  = (bf16*)alloc(BN * 3072 * 2);

    cvt_f32_to_bf16<<<512, 256, 0, stream>>>(qkv_w,  wqkv,  2304 * 768);
    cvt_f32_to_bf16<<<512, 256, 0, stream>>>(proj_w, wproj, 768 * 768);
    cvt_f32_to_bf16<<<512, 256, 0, stream>>>(fc1_w,  wfc1,  3072 * 768);
    cvt_f32_to_bf16<<<512, 256, 0, stream>>>(fc2_w,  wfc2,  768 * 3072);

    // LN1
    ln_768<<<(int)BN, 256, 0, stream>>>(x, ln1_g, ln1_b, hbf);
    // QKV projection: [BN,768] @ [2304,768]^T -> bf16 [BN,2304]
    gemm_bf16_wmma<0><<<dim3(2304 / 128, BN / 128), 256, 0, stream>>>(
        hbf, wqkv, nullptr, nullptr, qkvb, 2304, 768);
    // Fused attention -> bf16 [BN,768]
    attn_fused<<<dim3(1024 / 64, 8 * 12), 128, 0, stream>>>(qkvb, rel_t, aoutb);
    // proj + bias + residual(x) -> f32 y1
    gemm_bf16_wmma<1><<<dim3(768 / 128, BN / 128), 256, 0, stream>>>(
        aoutb, wproj, proj_b, x, y1, 768, 768);
    // LN2
    ln_768<<<(int)BN, 256, 0, stream>>>(y1, ln2_g, ln2_b, h2);
    // FC1 + bias + GELU -> bf16 [BN,3072]
    gemm_bf16_wmma<2><<<dim3(3072 / 128, BN / 128), 256, 0, stream>>>(
        h2, wfc1, fc1_b, nullptr, hbig, 3072, 768);
    // FC2 + bias + residual(y1) -> f32 d_out
    gemm_bf16_wmma<1><<<dim3(768 / 128, BN / 128), 256, 0, stream>>>(
        hbig, wfc2, fc2_b, y1, out, 768, 3072);
}
